// Mamba_53266184405022
// MI455X (gfx1250) — compile-verified, matrix operand order checked by reading, not run on hardware
//
#include <hip/hip_runtime.h>
#include <hip/hip_bf16.h>

typedef __attribute__((ext_vector_type(16))) __bf16 v16bf;
typedef __attribute__((ext_vector_type(2)))  __bf16 v2bf;
typedef __attribute__((ext_vector_type(8)))  float  v8f;

#define D_MODEL 768
#define D_INNER 1536
#define DSTATE  16
#define DTRANK  48
#define SEQB    2048          /* B * L */
#define SEQL    1024
#define NLAYER  4
#define VOCABN  32000
#define MIDP    128           /* padded x_proj output stride (was 80) */

#define BM 128
#define BN 128
#define BK 32

__device__ __forceinline__ unsigned short f2bf(float f) {
  unsigned int u = __float_as_uint(f);
  u += 0x7FFFu + ((u >> 16) & 1u);      // round-to-nearest-even
  return (unsigned short)(u >> 16);
}

__device__ __forceinline__ unsigned pack2bf(float lo, float hi) {
#if __has_builtin(__builtin_amdgcn_cvt_pk_bf16_f32)
  union { v2bf v; unsigned u; } c;
  c.v = __builtin_amdgcn_cvt_pk_bf16_f32(lo, hi);   // v_cvt_pk_bf16_f32
  return c.u;
#else
  return ((unsigned)f2bf(hi) << 16) | (unsigned)f2bf(lo);
#endif
}

union Frag { uint4 q[2]; v16bf v; };

// ---------------------------------------------------------------------------
// WMMA GEMM, fully compile-time shapes:  C[M,N] = act(A @ Bt^T + bias) + addsrc
//   A : f32 [M, LDA] (uses first K columns), M%128==0, N%128==0, K%32==0
//   Bt: bf16 [N, K]  (pre-converted, pre-transposed, zero-padded weights)
// Double-buffered LDS ping-pong:
//   B tile: global_load_async_to_lds_b128 (ASYNCcnt path, no VGPR round-trip)
//   A tile: f32 global_load_b128 -> v_cvt_pk_bf16_f32 -> ds_store_b128
//   tile kt+1 copy overlaps tile kt WMMA.
// Block = 256 threads = 8 waves; wave grid 4(M) x 2(N); wave tile 32x64.
// ---------------------------------------------------------------------------
template <int N, int K, int LDA, bool ADD, bool BIAS, bool SOFTPLUS>
__global__ __launch_bounds__(256)
void gemm_wmma_fast_kernel(const float* __restrict__ A,
                           const unsigned short* __restrict__ Bt,
                           float* __restrict__ C,
                           const float* __restrict__ addsrc,
                           const float* __restrict__ bias)
{
  __shared__ __align__(16) unsigned short As[2][BM][BK];   // [m][k] bf16
  __shared__ __align__(16) unsigned short Bs[2][BN][BK];   // [n][k] bf16

  const int tid  = threadIdx.x;
  const int lane = tid & 31;
  const int wave = tid >> 5;
  const int wm   = wave & 3;      // wave row (32 M-rows each)
  const int wn   = wave >> 2;     // wave col (64 N-cols each)
  const int r    = lane & 15;
  const int hi   = lane >> 4;
  const int hi8  = hi * 8;

  const int m0 = blockIdx.y * BM;
  const int n0 = blockIdx.x * BN;

  // staging assignments
  const int arow  = tid >> 1;            // 0..127
  const int acol  = (tid & 1) * 16;      // 0 or 16 (k offset)
  const int brn   = tid >> 1;            // 0..127
  const int bpart = (tid & 1) * 32;      // byte offset within 64B row

  const float*          Abase = A  + (size_t)(m0 + arow) * LDA + acol;
  const unsigned short* Bbase = Bt + (size_t)(n0 + brn) * K;
  const unsigned ldsB0 = (unsigned)(uintptr_t)&Bs[0][brn][0] + (unsigned)bpart;
  const unsigned ldsB1 = (unsigned)(uintptr_t)&Bs[1][brn][0] + (unsigned)bpart;

  const v8f vzero = {0.f,0.f,0.f,0.f,0.f,0.f,0.f,0.f};
  v8f acc[2][4];
  #pragma unroll
  for (int i = 0; i < 2; ++i)
    #pragma unroll
    for (int j = 0; j < 4; ++j) acc[i][j] = vzero;

  constexpr int nk = K / BK;

  auto stage = [&](int kt, int buf) {
    // B tile: async byte copy global -> LDS (2 x b128 per thread)
    const char* bsrc = (const char*)(Bbase + kt * BK) + bpart;
    unsigned lds = buf ? ldsB1 : ldsB0;
    asm volatile("global_load_async_to_lds_b128 %0, %1, off"
                 :: "v"(lds), "v"(bsrc) : "memory");
    asm volatile("global_load_async_to_lds_b128 %0, %1, off"
                 :: "v"(lds + 16u), "v"(bsrc + 16) : "memory");

    // A tile: f32 -> bf16 in registers, packed LDS stores
    const float* Ap = Abase + kt * BK;
    float4 f0 = *(const float4*)(Ap);
    float4 f1 = *(const float4*)(Ap + 4);
    float4 f2 = *(const float4*)(Ap + 8);
    float4 f3 = *(const float4*)(Ap + 12);
    uint4 p0, p1;
    p0.x = pack2bf(f0.x, f0.y);  p0.y = pack2bf(f0.z, f0.w);
    p0.z = pack2bf(f1.x, f1.y);  p0.w = pack2bf(f1.z, f1.w);
    p1.x = pack2bf(f2.x, f2.y);  p1.y = pack2bf(f2.z, f2.w);
    p1.z = pack2bf(f3.x, f3.y);  p1.w = pack2bf(f3.z, f3.w);
    *(uint4*)&As[buf][arow][acol]     = p0;
    *(uint4*)&As[buf][arow][acol + 8] = p1;
  };

  // ---- prologue: stage tile 0, wait, sync ----
  stage(0, 0);
  asm volatile("s_wait_asynccnt 0x0" ::: "memory");
  __syncthreads();

  for (int kt = 0; kt < nk; ++kt) {
    const int buf = kt & 1;

    // overlap: kick off copy of tile kt+1 into the other buffer
    if (kt + 1 < nk) stage(kt + 1, buf ^ 1);

    // ---- fragments + WMMA on tile kt ----
    Frag fa[2], fb[4];
    #pragma unroll
    for (int ms = 0; ms < 2; ++ms) {
      int row = wm * 32 + ms * 16 + r;
      fa[ms].q[0] = *(const uint4*)&As[buf][row][hi8];
      fa[ms].q[1] = *(const uint4*)&As[buf][row][16 + hi8];
    }
    #pragma unroll
    for (int ns = 0; ns < 4; ++ns) {
      int col = wn * 64 + ns * 16 + r;
      fb[ns].q[0] = *(const uint4*)&Bs[buf][col][hi8];
      fb[ns].q[1] = *(const uint4*)&Bs[buf][col][16 + hi8];
    }
    #pragma unroll
    for (int ms = 0; ms < 2; ++ms)
      #pragma unroll
      for (int ns = 0; ns < 4; ++ns)
        acc[ms][ns] = __builtin_amdgcn_wmma_f32_16x16x32_bf16(
            false, fa[ms].v, false, fb[ns].v, (short)0, acc[ms][ns],
            false, false);

    // tile kt+1's async copies must have landed before the next compute phase
    if (kt + 1 < nk)
      asm volatile("s_wait_asynccnt 0x0" ::: "memory");
    __syncthreads();
  }

  // ---- epilogue: C/D layout = VGPR v -> row (v + 8*hi), col = r ----
  #pragma unroll
  for (int ms = 0; ms < 2; ++ms) {
    #pragma unroll
    for (int ns = 0; ns < 4; ++ns) {
      const int gn  = n0 + wn * 64 + ns * 16 + r;
      const int gm0 = m0 + wm * 32 + ms * 16 + hi8;
      float bv = 0.f;
      if (BIAS) bv = bias[gn];
      float*       cp = C + (size_t)gm0 * N + gn;
      const float* ap = ADD ? (addsrc + (size_t)gm0 * N + gn) : nullptr;
      #pragma unroll
      for (int v = 0; v < 8; ++v) {
        float val = acc[ms][ns][v];
        if (BIAS)     val += bv;
        if (SOFTPLUS) val = (val > 20.f) ? val : log1pf(__expf(val));
        if (ADD)      val += ap[(size_t)v * N];
        cp[(size_t)v * N] = val;
      }
    }
  }
}

// ---------------------------------------------------------------------------
// Weight transform: f32 [Ksrc,Nsrc] -> bf16 [Npad,Kpad] transposed, zero-padded
// ---------------------------------------------------------------------------
__global__ __launch_bounds__(256)
void convT_pad_bf16_kernel(const float* __restrict__ W,
                           unsigned short* __restrict__ Wt,
                           int Nsrc, int Ksrc, int Kpad)
{
  int i  = blockIdx.x * 256 + threadIdx.x;   // over Npad*Kpad/2
  int kh = Kpad >> 1;
  int n  = i / kh;
  int k2 = (i % kh) * 2;
  float lo = (n < Nsrc && k2     < Ksrc) ? W[(size_t)k2 * Nsrc + n]       : 0.f;
  float hi = (n < Nsrc && k2 + 1 < Ksrc) ? W[(size_t)(k2 + 1) * Nsrc + n] : 0.f;
  ((unsigned*)Wt)[i] = pack2bf(lo, hi);
}

// straight f32 -> bf16 (embedding matrix, already [N,K] row-major)
__global__ __launch_bounds__(256)
void conv_bf16_kernel(const float* __restrict__ W, unsigned short* __restrict__ Wt,
                      int total2)                 // total/2 packed pairs
{
  int i = blockIdx.x * 256 + threadIdx.x;
  if (i >= total2) return;
  float2 f = *(const float2*)(W + (size_t)i * 2);
  ((unsigned*)Wt)[i] = pack2bf(f.x, f.y);
}

// ---------------------------------------------------------------------------
// Embedding gather:  x[m, :] = embed[ids[m], :]
// ---------------------------------------------------------------------------
__global__ __launch_bounds__(256)
void embed_kernel(const int* __restrict__ ids, const float* __restrict__ embed,
                  float* __restrict__ x)
{
  int i = blockIdx.x * 256 + threadIdx.x;       // SEQB*D_MODEL total
  int m = i / D_MODEL, c = i % D_MODEL;
  x[i] = embed[(size_t)ids[m] * D_MODEL + c];
}

// ---------------------------------------------------------------------------
// RMSNorm (scalar weight):  out = x * rsqrt(mean(x^2)+eps) * w[0]
// ---------------------------------------------------------------------------
__global__ __launch_bounds__(256)
void rmsnorm_kernel(const float* __restrict__ x, const float* __restrict__ w,
                    float* __restrict__ out)
{
  __shared__ float red[256];
  const int row = blockIdx.x, tid = threadIdx.x;
  const float* xr = x + (size_t)row * D_MODEL;
  float s = 0.f;
  for (int c = tid; c < D_MODEL; c += 256) { float v = xr[c]; s += v * v; }
  red[tid] = s;
  __syncthreads();
  for (int st = 128; st > 0; st >>= 1) {
    if (tid < st) red[tid] += red[tid + st];
    __syncthreads();
  }
  float scale = rsqrtf(red[0] * (1.f / D_MODEL) + 1e-15f) * w[0];
  for (int c = tid; c < D_MODEL; c += 256)
    out[(size_t)row * D_MODEL + c] = xr[c] * scale;
}

// ---------------------------------------------------------------------------
// Causal depthwise conv (K=4) + bias + SiLU, reading xc half of xr
// ---------------------------------------------------------------------------
__global__ __launch_bounds__(256)
void conv_silu_kernel(const float* __restrict__ xr, const float* __restrict__ cw,
                      const float* __restrict__ cb, float* __restrict__ xc)
{
  int i = blockIdx.x * 256 + threadIdx.x;       // SEQB*D_INNER total
  int m = i / D_INNER, d = i % D_INNER;
  int b = m >> 10, l = m & (SEQL - 1);
  float acc = cb[d];
  #pragma unroll
  for (int k = 0; k < 4; ++k) {
    int ll = l + k - 3;
    if (ll >= 0)
      acc += cw[k * D_INNER + d] *
             xr[((size_t)(b * SEQL + ll)) * (2 * D_INNER) + d];
  }
  xc[i] = acc / (1.f + __expf(-acc));           // silu
}

// ---------------------------------------------------------------------------
// Selective scan: one thread per (b, d) channel; 16 states in registers.
// B_/C vectors of the current timestep staged in LDS (shared by whole block).
// ---------------------------------------------------------------------------
__global__ __launch_bounds__(256)
void scan_kernel(const float* __restrict__ u, const float* __restrict__ delta,
                 const float* __restrict__ mid, const float* __restrict__ Alog,
                 const float* __restrict__ Dp, float* __restrict__ y)
{
  __shared__ float sB[DSTATE], sC[DSTATE];
  const int tid = threadIdx.x;
  const int b   = blockIdx.x / (D_INNER / 256);
  const int d   = (blockIdx.x % (D_INNER / 256)) * 256 + tid;

  float A[DSTATE];
  #pragma unroll
  for (int n = 0; n < DSTATE; ++n) A[n] = -__expf(Alog[(size_t)d * DSTATE + n]);
  const float Dv = Dp[d];

  float s[DSTATE];
  #pragma unroll
  for (int n = 0; n < DSTATE; ++n) s[n] = 0.f;

  for (int l = 0; l < SEQL; ++l) {
    size_t row = (size_t)b * SEQL + l;
    if (tid < DSTATE)            sB[tid]          = mid[row * MIDP + DTRANK + tid];
    else if (tid < 2 * DSTATE)   sC[tid - DSTATE] = mid[row * MIDP + DTRANK + DSTATE + (tid - DSTATE)];
    __syncthreads();

    float dl = delta[row * D_INNER + d];
    float uv = u[row * D_INNER + d];
    float du = dl * uv;
    float yv = 0.f;
    #pragma unroll
    for (int n = 0; n < DSTATE; ++n) {
      s[n] = __expf(dl * A[n]) * s[n] + du * sB[n];
      yv  += s[n] * sC[n];
    }
    y[row * D_INNER + d] = yv + uv * Dv;
    __syncthreads();
  }
}

// ---------------------------------------------------------------------------
// Gating:  y *= silu(res)  where res = xr[:, D_INNER:2*D_INNER]
// ---------------------------------------------------------------------------
__global__ __launch_bounds__(256)
void gate_kernel(float* __restrict__ y, const float* __restrict__ xr)
{
  int i = blockIdx.x * 256 + threadIdx.x;       // SEQB*D_INNER total
  int m = i / D_INNER, d = i % D_INNER;
  float rv = xr[(size_t)m * (2 * D_INNER) + D_INNER + d];
  y[i] *= rv / (1.f + __expf(-rv));
}

// ---------------------------------------------------------------------------
extern "C" void kernel_launch(void* const* d_in, const int* in_sizes, int n_in,
                              void* d_out, int out_size, void* d_ws, size_t ws_size,
                              hipStream_t stream)
{
  (void)in_sizes; (void)n_in; (void)out_size; (void)ws_size;

  const int*   ids   = (const int*)  d_in[0];
  const float* embed = (const float*)d_in[1];
  const float* in_w  = (const float*)d_in[2];
  const float* cw    = (const float*)d_in[3];
  const float* cb    = (const float*)d_in[4];
  const float* xw    = (const float*)d_in[5];
  const float* dtw   = (const float*)d_in[6];
  const float* dtb   = (const float*)d_in[7];
  const float* Alog  = (const float*)d_in[8];
  const float* Dp    = (const float*)d_in[9];
  const float* ow    = (const float*)d_in[10];
  const float* nw    = (const float*)d_in[11];
  const float* nfw   = (const float*)d_in[12];
  float* logits = (float*)d_out;

  float* ws = (float*)d_ws;
  size_t o = 0;
  float* x     = ws + o;  o += (size_t)SEQB * D_MODEL;        // residual stream
  float* xn    = ws + o;  o += (size_t)SEQB * D_MODEL;        // normed
  float* xr    = ws + o;  o += (size_t)SEQB * 2 * D_INNER;    // in_proj out
  float* xc    = ws + o;  o += (size_t)SEQB * D_INNER;        // conv+silu out
  float* mid   = ws + o;  o += (size_t)SEQB * MIDP;           // x_proj out (padded)
  float* delta = ws + o;  o += (size_t)SEQB * D_INNER;        // softplus(dt)
  float* ybuf  = ws + o;  o += (size_t)SEQB * D_INNER;        // scan out
  unsigned short* wbf = (unsigned short*)(ws + o);            // bf16 weight scratch
                                                              // (max 32000*768 el)

  const dim3 blk(256);

  embed_kernel<<<(SEQB * D_MODEL) / 256, blk, 0, stream>>>(ids, embed, x);

  for (int layer = 0; layer < NLAYER; ++layer) {
    rmsnorm_kernel<<<SEQB, blk, 0, stream>>>(x, nw + layer, xn);

    // in_proj: [2048,768] @ [768,3072]  (weights -> bf16 [3072,768])
    convT_pad_bf16_kernel<<<(2 * D_INNER * D_MODEL / 2) / 256, blk, 0, stream>>>(
        in_w + (size_t)layer * D_MODEL * 2 * D_INNER, wbf,
        2 * D_INNER, D_MODEL, D_MODEL);
    gemm_wmma_fast_kernel<2 * D_INNER, D_MODEL, D_MODEL, false, false, false>
        <<<dim3((2 * D_INNER) / BN, SEQB / BM), blk, 0, stream>>>(
            xn, wbf, xr, nullptr, nullptr);

    conv_silu_kernel<<<(SEQB * D_INNER) / 256, blk, 0, stream>>>(
        xr, cw + (size_t)layer * 4 * D_INNER, cb + (size_t)layer * D_INNER, xc);

    // x_proj: [2048,1536] @ [1536,80] -> N padded to 128 (weight rows 80.. zero)
    convT_pad_bf16_kernel<<<(MIDP * D_INNER / 2) / 256, blk, 0, stream>>>(
        xw + (size_t)layer * D_INNER * 80, wbf, 80, D_INNER, D_INNER);
    gemm_wmma_fast_kernel<MIDP, D_INNER, D_INNER, false, false, false>
        <<<dim3(MIDP / BN, SEQB / BM), blk, 0, stream>>>(
            xc, wbf, mid, nullptr, nullptr);

    // dt_proj + softplus: [2048,48] @ [48,1536] + dtb
    //   K padded 48 -> 64 (zero weight rows), A = mid[:, 0:64] with LDA=128
    convT_pad_bf16_kernel<<<(D_INNER * 64 / 2) / 256, blk, 0, stream>>>(
        dtw + (size_t)layer * DTRANK * D_INNER, wbf, D_INNER, DTRANK, 64);
    gemm_wmma_fast_kernel<D_INNER, 64, MIDP, false, true, true>
        <<<dim3(D_INNER / BN, SEQB / BM), blk, 0, stream>>>(
            mid, wbf, delta, nullptr, dtb + (size_t)layer * D_INNER);

    scan_kernel<<<2 * (D_INNER / 256), blk, 0, stream>>>(
        xc, delta, mid, Alog + (size_t)layer * D_INNER * DSTATE,
        Dp + (size_t)layer * D_INNER, ybuf);

    gate_kernel<<<(SEQB * D_INNER) / 256, blk, 0, stream>>>(ybuf, xr);

    // out_proj + residual: [2048,1536] @ [1536,768] + x -> x
    convT_pad_bf16_kernel<<<(D_MODEL * D_INNER / 2) / 256, blk, 0, stream>>>(
        ow + (size_t)layer * D_INNER * D_MODEL, wbf, D_MODEL, D_INNER, D_INNER);
    gemm_wmma_fast_kernel<D_MODEL, D_INNER, D_INNER, true, false, false>
        <<<dim3(D_MODEL / BN, SEQB / BM), blk, 0, stream>>>(
            ybuf, wbf, x, x, nullptr);
  }

  rmsnorm_kernel<<<SEQB, blk, 0, stream>>>(x, nfw, xn);

  // logits: [2048,768] @ embed^T;  embed is [32000,768] == already [N,K]
  conv_bf16_kernel<<<(VOCABN * D_MODEL / 2) / 256, blk, 0, stream>>>(
      embed, wbf, (VOCABN * D_MODEL) / 2);
  gemm_wmma_fast_kernel<VOCABN, D_MODEL, D_MODEL, false, false, false>
      <<<dim3(VOCABN / BN, SEQB / BM), blk, 0, stream>>>(
          xn, wbf, logits, nullptr, nullptr);
}
